// StarPooling_3015067041829
// MI455X (gfx1250) — compile-verified, matrix operand order checked by reading, not run on hardware
//
#include <hip/hip_runtime.h>
#include <hip/hip_bf16.h>

#define N_NODES 4096
#define N_EDGES 65536
#define D_FEAT  64

typedef __attribute__((ext_vector_type(16))) _Float16 v16h;
typedef __attribute__((ext_vector_type(8)))  float    v8f;

// ---------------------------------------------------------------------------
// Init: zero agg, copy x -> x_work, copy edge_index -> working src/dst
// ---------------------------------------------------------------------------
__global__ void k_init(const float* __restrict__ x, const int* __restrict__ ei,
                       float* agg, float* x_work, int* srcW, int* dstW) {
  int stride = gridDim.x * blockDim.x;
  int i = blockIdx.x * blockDim.x + threadIdx.x;
  const int total = N_NODES * D_FEAT;
  for (int idx = i; idx < total; idx += stride) {
    agg[idx] = 0.0f;
    x_work[idx] = x[idx];
  }
  for (int e = i; e < N_EDGES; e += stride) {
    srcW[e] = ei[e];
    dstW[e] = ei[N_EDGES + e];
  }
}

// ---------------------------------------------------------------------------
// Aggregation: agg[dst] += x[src] (one thread per edge x feature)
// ---------------------------------------------------------------------------
__global__ void k_agg(const float* __restrict__ x, const int* __restrict__ ei,
                      float* agg) {
  int idx = blockIdx.x * blockDim.x + threadIdx.x;
  if (idx >= N_EDGES * D_FEAT) return;
  int e = idx >> 6;
  int f = idx & 63;
  int s = ei[e];
  int d = ei[N_EDGES + e];
  atomicAdd(&agg[d * D_FEAT + f], x[s * D_FEAT + f]);
}

// ---------------------------------------------------------------------------
// Score GEMV via WMMA: score = tanh(agg @ w_rel + x @ w_root + b)
// One wave per 16 rows; w broadcast to all 16 B columns so every lane's C
// VGPRs hold the row dot-products (C/D layout: lane L, vgpr v -> M = 8*(L>>4)+v).
// ---------------------------------------------------------------------------
__device__ inline v16h load_a_tile(const float* __restrict__ M, int m0, int r,
                                   int hi, int kk) {
  // 16-bit A 16x32 layout: lane (r, hi); vgpr v<4 -> K = 2v + 8*hi (+pos),
  // vgpr v>=4 -> K = 16 + 2(v-4) + 8*hi (+pos)
  v16h a;
#pragma unroll
  for (int j = 0; j < 16; ++j) {
    int v = j >> 1, pos = j & 1;
    int base = (v < 4) ? (2 * v) : (16 + 2 * (v - 4));
    int K = kk + base + 8 * hi + pos;
    a[j] = (_Float16)M[(m0 + r) * D_FEAT + K];
  }
  return a;
}

__device__ inline v16h load_b_bcast(const float* __restrict__ w, int hi, int kk) {
  // B[k][n] = w[k] for every column n (mirror of A's K mapping per lane)
  v16h b;
#pragma unroll
  for (int j = 0; j < 16; ++j) {
    int v = j >> 1, pos = j & 1;
    int base = (v < 4) ? (2 * v) : (16 + 2 * (v - 4));
    int K = kk + base + 8 * hi + pos;
    b[j] = (_Float16)w[K];
  }
  return b;
}

__global__ void k_score(const float* __restrict__ agg, const float* __restrict__ x,
                        const float* __restrict__ w_rel,
                        const float* __restrict__ w_root,
                        const float* __restrict__ bptr, float* score) {
  int lane = threadIdx.x;
  int r = lane & 15;
  int hi = lane >> 4;
  int m0 = blockIdx.x * 16;

  v8f c = {};
  {
    v16h a = load_a_tile(agg, m0, r, hi, 0);
    v16h b = load_b_bcast(w_rel, hi, 0);
    c = __builtin_amdgcn_wmma_f32_16x16x32_f16(false, a, false, b, (short)0, c, false, false);
  }
  {
    v16h a = load_a_tile(agg, m0, r, hi, 32);
    v16h b = load_b_bcast(w_rel, hi, 32);
    c = __builtin_amdgcn_wmma_f32_16x16x32_f16(false, a, false, b, (short)0, c, false, false);
  }
  {
    v16h a = load_a_tile(x, m0, r, hi, 0);
    v16h b = load_b_bcast(w_root, hi, 0);
    c = __builtin_amdgcn_wmma_f32_16x16x32_f16(false, a, false, b, (short)0, c, false, false);
  }
  {
    v16h a = load_a_tile(x, m0, r, hi, 32);
    v16h b = load_b_bcast(w_root, hi, 32);
    c = __builtin_amdgcn_wmma_f32_16x16x32_f16(false, a, false, b, (short)0, c, false, false);
  }

  if (r == 0) {  // lanes 0 and 16 cover rows 0..7 and 8..15 via vgprs 0..7
    float bias = bptr[0];
#pragma unroll
    for (int v = 0; v < 8; ++v) {
      int m = hi * 8 + v;
      score[m0 + m] = tanhf(c[v] + bias);
    }
  }
}

// ---------------------------------------------------------------------------
// Stable descending rank (argsort of -score with index tie-break)
// ---------------------------------------------------------------------------
__global__ void k_rank(const float* __restrict__ score, int* order) {
  int i = blockIdx.x * blockDim.x + threadIdx.x;
  if (i >= N_NODES) return;
  float si = score[i];
  int rank = 0;
  for (int j = 0; j < N_NODES; ++j) {
    float sj = score[j];
    rank += (sj > si) || (sj == si && j < i);
  }
  order[rank] = i;
}

// ---------------------------------------------------------------------------
// Sequential greedy star contraction: single persistent workgroup.
// Hot flags live in LDS; mutated global arrays have fixed per-thread element
// ownership (edge e -> thread e%1024, feature column f -> thread f), so no
// cross-CU L0 coherence hazards exist inside the loop.
// ---------------------------------------------------------------------------
__global__ void __launch_bounds__(1024)
k_merge(const int* __restrict__ order, float* x_work, int* srcW, int* dstW,
        int* remaining_g) {
  __shared__ int list[N_NODES];               // compacted dest set
  __shared__ unsigned char destF[N_NODES];
  __shared__ unsigned char rem[N_NODES];
  __shared__ unsigned char cen[N_NODES];
  __shared__ int s_cnt, s_node, s_active;

  const int tid = threadIdx.x;
  const int BS = blockDim.x;

  for (int n = tid; n < N_NODES; n += BS) { rem[n] = 1; cen[n] = 0; }
  __syncthreads();

  for (int it = 0; it < N_NODES; ++it) {
    if (tid == 0) {
      int nd = order[it];
      s_node = nd;
      s_active = rem[nd];
      s_cnt = 0;
    }
    __syncthreads();
    if (s_active) {
      const int node = s_node;
      // Phase 1: clear dest flags
      for (int n = tid; n < N_NODES; n += BS) destF[n] = 0;
      __syncthreads();
      // Phase 2: mark dest nodes of edges with (rewritten) source == node
      for (int e = tid; e < N_EDGES; e += BS)
        if (srcW[e] == node) destF[dstW[e]] = 1;
      __syncthreads();
      // Phase 3: mask by ~center, drop from remaining, compact into list
      for (int n = tid; n < N_NODES; n += BS) {
        if (destF[n] && !cen[n]) {
          rem[n] = 0;
          int p = atomicAdd(&s_cnt, 1);
          list[p] = n;
        }
      }
      __syncthreads();
      // Phase 4a: combine = sum(dest rows) + x[node]  (column-owner threads)
      if (tid < D_FEAT) {
        float acc = x_work[node * D_FEAT + tid];
        int c = s_cnt;
        for (int i = 0; i < c; ++i) acc += x_work[list[i] * D_FEAT + tid];
        x_work[node * D_FEAT + tid] = acc;
      }
      // Phase 4b (concurrent, independent): rewrite edges with D = dest U {node}
      for (int e = tid; e < N_EDGES; e += BS) {
        int s = srcW[e];
        if ((destF[s] && !cen[s]) || s == node) srcW[e] = node;
        int d = dstW[e];
        if ((destF[d] && !cen[d]) || d == node) dstW[e] = node;
      }
      __syncthreads();
      if (tid == 0) cen[node] = 1;
    }
    __syncthreads();
  }

  for (int n = tid; n < N_NODES; n += BS) remaining_g[n] = rem[n];
}

// ---------------------------------------------------------------------------
// relabel = cumsum(remaining)-1 ; perm initialized to sentinel N
// ---------------------------------------------------------------------------
__global__ void k_relabel(const int* __restrict__ remaining, int* relabel,
                          int* perm) {
  int i = blockIdx.x * blockDim.x + threadIdx.x;
  if (i >= N_NODES) return;
  int c = 0;
  for (int j = 0; j <= i; ++j) c += remaining[j];
  relabel[i] = c - 1;
  perm[i] = N_NODES;
}

__global__ void k_scatter(const int* __restrict__ remaining,
                          const int* __restrict__ relabel, int* perm) {
  int i = blockIdx.x * blockDim.x + threadIdx.x;
  if (i >= N_NODES) return;
  if (remaining[i]) perm[relabel[i]] = i;
}

// ---------------------------------------------------------------------------
// Outputs: [x_pooled (N*64)] [new_edge_index (2*E)] [batch_pooled (N)] [perm (N)]
// all written numerically as f32 into the concatenated d_out.
// ---------------------------------------------------------------------------
__global__ void k_out(const float* __restrict__ x_work, const int* __restrict__ ei,
                      const int* __restrict__ batch,
                      const int* __restrict__ remaining,
                      const int* __restrict__ relabel,
                      const int* __restrict__ perm, float* out) {
  const int OUT_X = N_NODES * D_FEAT;
  const int OUT_E = 2 * N_EDGES;
  const int total = OUT_X + OUT_E + 2 * N_NODES;
  int idx = blockIdx.x * blockDim.x + threadIdx.x;
  if (idx >= total) return;

  if (idx < OUT_X) {
    int k = idx / D_FEAT, f = idx % D_FEAT;
    int p = perm[k];
    out[idx] = (p < N_NODES) ? x_work[p * D_FEAT + f] : 0.0f;
  } else if (idx < OUT_X + OUT_E) {
    int t = idx - OUT_X;
    int row = t / N_EDGES;  // 0 = src, 1 = dst
    int e = t % N_EDGES;
    int s = ei[e], d = ei[N_EDGES + e];
    bool valid = remaining[s] && remaining[d];
    int val = valid ? relabel[(row == 0) ? s : d] : -1;
    out[idx] = (float)val;
  } else if (idx < OUT_X + OUT_E + N_NODES) {
    int i = idx - OUT_X - OUT_E;
    int p = perm[i];
    out[idx] = (p < N_NODES) ? (float)batch[p] : -1.0f;
  } else {
    int i = idx - OUT_X - OUT_E - N_NODES;
    int p = perm[i];
    out[idx] = (p < N_NODES) ? (float)p : -1.0f;
  }
}

// ---------------------------------------------------------------------------
extern "C" void kernel_launch(void* const* d_in, const int* in_sizes, int n_in,
                              void* d_out, int out_size, void* d_ws, size_t ws_size,
                              hipStream_t stream) {
  const float* x      = (const float*)d_in[0];
  const int*   ei     = (const int*)d_in[1];
  const int*   batch  = (const int*)d_in[2];
  const float* w_rel  = (const float*)d_in[3];
  const float* w_root = (const float*)d_in[4];
  const float* b      = (const float*)d_in[5];
  float* out = (float*)d_out;

  // workspace layout (f32/i32 elements, ~2.7 MB total)
  float* agg       = (float*)d_ws;
  float* x_work    = agg + N_NODES * D_FEAT;
  float* score     = x_work + N_NODES * D_FEAT;
  int*   order     = (int*)(score + N_NODES);
  int*   srcW      = order + N_NODES;
  int*   dstW      = srcW + N_EDGES;
  int*   remaining = dstW + N_EDGES;
  int*   relabel   = remaining + N_NODES;
  int*   perm      = relabel + N_NODES;

  k_init<<<256, 256, 0, stream>>>(x, ei, agg, x_work, srcW, dstW);
  k_agg<<<(N_EDGES * D_FEAT + 255) / 256, 256, 0, stream>>>(x, ei, agg);
  k_score<<<N_NODES / 16, 32, 0, stream>>>(agg, x, w_rel, w_root, b, score);
  k_rank<<<(N_NODES + 255) / 256, 256, 0, stream>>>(score, order);
  k_merge<<<1, 1024, 0, stream>>>(order, x_work, srcW, dstW, remaining);
  k_relabel<<<(N_NODES + 255) / 256, 256, 0, stream>>>(remaining, relabel, perm);
  k_scatter<<<(N_NODES + 255) / 256, 256, 0, stream>>>(remaining, relabel, perm);
  const int total = N_NODES * D_FEAT + 2 * N_EDGES + 2 * N_NODES;
  k_out<<<(total + 255) / 256, 256, 0, stream>>>(x_work, ei, batch, remaining,
                                                 relabel, perm, out);
}
